// SelfAttention_41961830482241
// MI455X (gfx1250) — compile-verified, matrix operand order checked by reading, not run on hardware
//
#include <hip/hip_runtime.h>
#include <stdint.h>

// MI455X / gfx1250, wave32. Fused flash-attention for the reference:
//   f=wq@x (keys), g=wk@x (queries), h=wv@x (values), head dim 8, N=4096, B=4.
//   out = gamma * (h @ softmax_n(f^T g)) + x
// Matmuls on v_wmma_f32_16x16x32_bf16 (K=32, head-dim 8 zero-padded).

typedef __attribute__((ext_vector_type(16))) __bf16 v16bf;
typedef __attribute__((ext_vector_type(8)))  float  v8f;

#define N_TOK 4096
#define CCH   64
#define KD16  16     // padded head-dim slots per token (8 real + 8 zero)
#define BATCH 4

union FragB { v16bf v; uint32_t u[8]; };

// ---------------------------------------------------------------------------
// Kernel 1: projections f,g,h  (f32 in -> bf16 out), x tile staged in LDS.
//   fT: [B][N][16]  keys,    k=0..7 real, k=8..15 zero (fragment padding)
//   gT: [B][N][16]  queries, same padding
//   hB: [B][64][N]  values, channel-major so PV A-fragments are 16B loads
// ---------------------------------------------------------------------------
__global__ __launch_bounds__(256) void SelfAttn_proj_kernel(
    const float* __restrict__ x,  const float* __restrict__ wq,
    const float* __restrict__ wk, const float* __restrict__ wv,
    __bf16* __restrict__ fT, __bf16* __restrict__ gT,
    __bf16* __restrict__ hB)
{
    __shared__ float xs[64][65];          // [n][c], padded to kill bank conflicts
    const int b   = blockIdx.x >> 6;      // 64 token-tiles per batch
    const int nt  = (blockIdx.x & 63) << 6;
    const int tid = threadIdx.x;
    const int n   = tid & 63;
    const int r0  = tid >> 6;             // 0..3

    #pragma unroll
    for (int i = 0; i < 16; ++i) {
        const int c = r0 + i * 4;
        xs[n][c] = x[((size_t)b * CCH + c) * N_TOK + nt + n];  // coalesced in n
    }

    // zero-fill the k=8..15 pad slots for this token tile
    for (int i = tid; i < 64 * 8; i += 256) {
        const int tok = nt + (i >> 3);
        const int k   = 8 + (i & 7);
        fT[((size_t)b * N_TOK + tok) * KD16 + k] = (__bf16)0.f;
        gT[((size_t)b * N_TOK + tok) * KD16 + k] = (__bf16)0.f;
    }
    __syncthreads();

    for (int rr = r0; rr < 80; rr += 4) {   // rows: 0-7 f, 8-15 g, 16-79 h
        const float* wrow;
        if (rr < 8)       wrow = wq + rr * CCH;
        else if (rr < 16) wrow = wk + (rr - 8) * CCH;
        else              wrow = wv + (rr - 16) * CCH;
        float acc = 0.f;
        #pragma unroll 8
        for (int c = 0; c < CCH; ++c) acc += wrow[c] * xs[n][c];
        const __bf16 bv = (__bf16)acc;       // native RNE cvt
        const int tok = nt + n;
        if (rr < 8)       fT[((size_t)b * N_TOK + tok) * KD16 + rr] = bv;
        else if (rr < 16) gT[((size_t)b * N_TOK + tok) * KD16 + (rr - 8)] = bv;
        else              hB[((size_t)b * CCH + (rr - 16)) * N_TOK + tok] = bv;
    }
}

// ---------------------------------------------------------------------------
// Kernel 2: fused flash attention. One wave32 per 16-query tile.
//   Per iteration: 32 keys -> 2 score WMMAs + online softmax + 4 PV WMMAs.
// ---------------------------------------------------------------------------
__global__ __launch_bounds__(256) void SelfAttn_flash_kernel(
    const uint16_t* __restrict__ fT,   // keys   [B][N][16] (padded)
    const uint16_t* __restrict__ gT,   // queries[B][N][16] (padded)
    const uint16_t* __restrict__ hB,   // values [B][64][N]
    const float* __restrict__ x,
    const float* __restrict__ gamma,
    float* __restrict__ out)
{
    const int lane  = threadIdx.x & 31;
    const int gw    = blockIdx.x * 8 + (threadIdx.x >> 5);  // global wave id
    const int b     = gw >> 8;                              // 256 m-tiles/batch
    const int m0    = (gw & 255) << 4;
    const int lm    = lane & 15;
    const int khalf = (lane >> 4) << 3;   // lanes 0-15 -> K 0..7, 16-31 -> K 8..15

    // Query fragment (WMMA B, 32x16): col m = lane%16. K=8..31 are zeros
    // (k=8..15 read from the padded store, k=16..31 zeroed in registers).
    FragB qf;
    {
        const uint4 q = *(const uint4*)(gT + ((size_t)b * N_TOK + m0 + lm) * KD16 + khalf);
        qf.u[0] = q.x; qf.u[1] = q.y; qf.u[2] = q.z; qf.u[3] = q.w;
        qf.u[4] = 0u; qf.u[5] = 0u; qf.u[6] = 0u; qf.u[7] = 0u;
    }

    v8f oacc[4];
    #pragma unroll
    for (int ct = 0; ct < 4; ++ct) oacc[ct] = (v8f){};
    float run_max = -3.0e38f;
    float run_sum = 0.f;

    const uint16_t* fbase = fT + (size_t)b * N_TOK * KD16;
    const uint16_t* hbase = hB + (size_t)b * CCH * N_TOK;

    for (int nt = 0; nt < N_TOK; nt += 32) {
        // --- two key tiles (WMMA A, 16x32): row n = lane%16; lanes 0-15 load
        //     K=0..7 (real), lanes 16-31 load K=8..15 (stored zeros).
        FragB a0, a1;
        {
            const uint4 k0 = *(const uint4*)(fbase + (size_t)(nt + lm) * KD16 + khalf);
            const uint4 k1 = *(const uint4*)(fbase + (size_t)(nt + 16 + lm) * KD16 + khalf);
            a0.u[0] = k0.x; a0.u[1] = k0.y; a0.u[2] = k0.z; a0.u[3] = k0.w;
            a0.u[4] = 0u; a0.u[5] = 0u; a0.u[6] = 0u; a0.u[7] = 0u;
            a1.u[0] = k1.x; a1.u[1] = k1.y; a1.u[2] = k1.z; a1.u[3] = k1.w;
            a1.u[4] = 0u; a1.u[5] = 0u; a1.u[6] = 0u; a1.u[7] = 0u;
        }
        v8f s0 = (v8f){}, s1 = (v8f){};
        s0 = __builtin_amdgcn_wmma_f32_16x16x32_bf16(false, a0.v, false, qf.v,
                                                     (short)0, s0, false, false);
        s1 = __builtin_amdgcn_wmma_f32_16x16x32_bf16(false, a1.v, false, qf.v,
                                                     (short)0, s1, false, false);

        // --- online softmax over n (rows): in-lane reduce + xor-16 partner
        float tmax = s0[0];
        #pragma unroll
        for (int i = 1; i < 8; ++i) tmax = fmaxf(tmax, s0[i]);
        #pragma unroll
        for (int i = 0; i < 8; ++i) tmax = fmaxf(tmax, s1[i]);
        tmax = fmaxf(tmax, __shfl_xor(tmax, 16, 32));
        const float nmax  = fmaxf(run_max, tmax);
        const float scale = __expf(run_max - nmax);
        run_max = nmax;

        float p0[8], p1[8];
        float tsum = 0.f;
        #pragma unroll
        for (int i = 0; i < 8; ++i) { p0[i] = __expf(s0[i] - nmax); tsum += p0[i]; }
        #pragma unroll
        for (int i = 0; i < 8; ++i) { p1[i] = __expf(s1[i] - nmax); tsum += p1[i]; }
        tsum += __shfl_xor(tsum, 16, 32);
        run_sum = run_sum * scale + tsum;

        #pragma unroll
        for (int ct = 0; ct < 4; ++ct) oacc[ct] *= scale;

        // --- P fragment (WMMA B, 32x16): S-tile D layout maps directly onto
        //     B-fragment element slots; native f32->bf16 converts, no shuffles.
        FragB pf;
        #pragma unroll
        for (int i = 0; i < 8; ++i) {
            pf.v[i]     = (__bf16)p0[i];   // K 0..7  (lo half) / 8..15  (hi half)
            pf.v[8 + i] = (__bf16)p1[i];   // K 16..23 (lo)     / 24..31 (hi)
        }

        // --- PV: O[c,m] += h[c,n] * P[n,m], 4 channel tiles of 16
        const int nlo = nt + khalf;        // lo half: nt, hi half: nt+8
        #pragma unroll
        for (int ct = 0; ct < 4; ++ct) {
            FragB hf;
            const uint16_t* hr = hbase + (size_t)(ct * 16 + lm) * N_TOK;
            const uint4 h0 = *(const uint4*)(hr + nlo);        // K 0..7  (or 8..15)
            const uint4 h1 = *(const uint4*)(hr + nlo + 16);   // K 16..23 (or 24..31)
            hf.u[0] = h0.x; hf.u[1] = h0.y; hf.u[2] = h0.z; hf.u[3] = h0.w;
            hf.u[4] = h1.x; hf.u[5] = h1.y; hf.u[6] = h1.z; hf.u[7] = h1.w;
            __builtin_prefetch(hr + nlo + 64, 0, 1);           // 2 tiles ahead
            oacc[ct] = __builtin_amdgcn_wmma_f32_16x16x32_bf16(
                false, hf.v, false, pf.v, (short)0, oacc[ct], false, false);
        }
    }

    // --- epilogue: normalize, gamma, residual
    const float g0  = gamma[0];
    const float inv = 1.0f / run_sum;
    const int   m   = m0 + lm;
    #pragma unroll
    for (int ct = 0; ct < 4; ++ct) {
        #pragma unroll
        for (int v = 0; v < 8; ++v) {
            const int c = ct * 16 + v + khalf;
            const size_t idx = ((size_t)b * CCH + c) * N_TOK + m;
            out[idx] = g0 * (oacc[ct][v] * inv) + x[idx];
        }
    }
}

extern "C" void kernel_launch(void* const* d_in, const int* in_sizes, int n_in,
                              void* d_out, int out_size, void* d_ws, size_t ws_size,
                              hipStream_t stream) {
    (void)in_sizes; (void)n_in; (void)out_size; (void)ws_size;
    const float* x     = (const float*)d_in[0];
    const float* wq    = (const float*)d_in[1];
    const float* wk    = (const float*)d_in[2];
    const float* wv    = (const float*)d_in[3];
    const float* gamma = (const float*)d_in[4];
    float* out = (float*)d_out;

    uint16_t* fT = (uint16_t*)d_ws;                          // B*N*16 bf16
    uint16_t* gT = fT + (size_t)BATCH * N_TOK * KD16;        // B*N*16 bf16
    uint16_t* hB = gT + (size_t)BATCH * N_TOK * KD16;        // B*64*N bf16
    // total workspace: (2*4*4096*16 + 4*64*4096) * 2B ~= 3.1 MB

    SelfAttn_proj_kernel<<<BATCH * (N_TOK / 64), 256, 0, stream>>>(
        x, wq, wk, wv, (__bf16*)fT, (__bf16*)gT, (__bf16*)hB);
    // 1024 query-tile waves, 8 waves per block
    SelfAttn_flash_kernel<<<BATCH * (N_TOK / 16) / 8, 256, 0, stream>>>(
        fT, gT, hB, x, gamma, out);
}
